// CTGRU_19593640804384
// MI455X (gfx1250) — matrix operand order repaired and checked by reading, not compiled, not run on hardware
//
#include <hip/hip_runtime.h>
#include <hip/hip_bf16.h>

#define B_ROWS 8192
#define D_IN   256
#define UNITS  512
#define MBUCK  8
#define KDIM   768      // D_IN + UNITS
#define NGATE  4096     // UNITS * M
#define KSTEPS (KDIM / 32)
#define LDS_ROW 80      // 64B of K data + 16B TDM pad -> conflict-free ds reads
#define LDS_BUF (64 * LDS_ROW)   // 5120 B per k-step panel

typedef __bf16 bf16_t;
typedef __attribute__((ext_vector_type(8)))  bf16_t v8bf;
typedef __attribute__((ext_vector_type(16))) bf16_t v16bf;
typedef __attribute__((ext_vector_type(8)))  float  v8f;
typedef __attribute__((ext_vector_type(4)))  unsigned int u32x4;
typedef __attribute__((ext_vector_type(4)))  int  i32x4;
typedef __attribute__((ext_vector_type(8)))  int  i32x8;

// ---------------------------------------------------------------------------
// TDM: DMA one 32(K) x 64(N) bf16 tile of the N-major weight panel into LDS.
// D# per CDNA5 ISA 8.3-8.6: count=1, type=2, data_size=2B, pad every 16 dwords
// by 4 dwords => LDS row stride 80B (bank-conflict-free fragment reads).
// ---------------------------------------------------------------------------
__device__ inline void tdm_load_panel(const bf16_t* gsrc, unsigned lds_off,
                                      int nrows) {
  const unsigned long long ga = (unsigned long long)(uintptr_t)gsrc;
  u32x4 g0;
  g0[0] = 1u;                                    // count=1 (valid descriptor)
  g0[1] = lds_off;                               // lds_addr
  g0[2] = (unsigned)(ga & 0xffffffffull);        // global_addr[31:0]
  g0[3] = (unsigned)((ga >> 32) & 0x1ffffffull)  // global_addr[56:32]
          | 0x80000000u;                         // type=2 ("image")
  i32x8 g1;
  g1[0] = (1 << 16)        // data_size = 2 bytes
        | (1 << 20)        // pad_enable
        | (3 << 22)        // pad_interval = 16 dwords (64B)
        | (3 << 25);       // pad_amount   = 4 dwords (16B)
  g1[1] = (int)((unsigned)KDIM << 16);           // tensor_dim0 = 768 (lo16)
  g1[2] = (int)((unsigned)nrows << 16);          // tensor_dim1 lo16
  g1[3] = (int)(32u << 16);                      // tile_dim0 = 32
  g1[4] = 64;                                    // tile_dim1 = 64, tile_dim2 = 0
  g1[5] = KDIM;                                  // tensor_dim0_stride
  g1[6] = 0;
  g1[7] = 0;
  i32x4 g2; g2[0] = 1; g2[1] = 1; g2[2] = 0; g2[3] = 0;
  i32x4 g3; g3[0] = 0; g3[1] = 0; g3[2] = 0; g3[3] = 0;
#if __clang_major__ >= 23
  i32x8 g4; for (int i = 0; i < 8; ++i) g4[i] = 0;
  __builtin_amdgcn_tensor_load_to_lds(g0, g1, g2, g3, g4, 0);
#else
  __builtin_amdgcn_tensor_load_to_lds(g0, g1, g2, g3, 0);
#endif
}

__device__ inline v16bf pack16(v8bf lo, v8bf hi) {
  v16bf r;
  #pragma unroll
  for (int i = 0; i < 8; ++i) { r[i] = lo[i]; r[i + 8] = hi[i]; }
  return r;
}

// ---------------------------------------------------------------------------
// Transpose + f32->bf16 convert:  in f32 [K][N] row-major  ->  out bf16 [N][K]
// ---------------------------------------------------------------------------
__global__ void k_transpose_cvt(const float* __restrict__ in,
                                bf16_t* __restrict__ out, int K, int N) {
  __shared__ float tile[32][33];
  const int nBase = blockIdx.x * 32;
  const int kBase = blockIdx.y * 32;
  #pragma unroll 4
  for (int j = threadIdx.y; j < 32; j += 8) {
    tile[j][threadIdx.x] = in[(size_t)(kBase + j) * N + nBase + threadIdx.x];
  }
  __syncthreads();
  #pragma unroll 4
  for (int j = threadIdx.y; j < 32; j += 8) {
    out[(size_t)(nBase + j) * K + kBase + threadIdx.x] = (bf16_t)tile[threadIdx.x][j];
  }
}

// ---------------------------------------------------------------------------
// fused = [bf16(inputs) | bf16(sum_m h_hat)]  (B x 768); also seed reset cols.
// ---------------------------------------------------------------------------
__global__ void k_fuse(const float* __restrict__ inputs,
                       const float* __restrict__ state,
                       bf16_t* __restrict__ fused,
                       bf16_t* __restrict__ reset) {
  const int b = blockIdx.x;
  const int t = threadIdx.x;
  const bf16_t xb = (bf16_t)inputs[(size_t)b * D_IN + t];
  fused[(size_t)b * KDIM + t] = xb;
  reset[(size_t)b * KDIM + t] = xb;
  #pragma unroll
  for (int uu = 0; uu < 2; ++uu) {
    const int u = t + uu * 256;
    const float4* sp = (const float4*)(state + (size_t)b * NGATE + u * MBUCK);
    const float4 s0 = sp[0];
    const float4 s1 = sp[1];
    const float sum = s0.x + s0.y + s0.z + s0.w + s1.x + s1.y + s1.z + s1.w;
    fused[(size_t)b * KDIM + D_IN + u] = (bf16_t)sum;
  }
}

// ---------------------------------------------------------------------------
// bf16 WMMA GEMM, B panel staged through LDS by the TDM (double buffered).
//   MODE 0: retrieval gate -> softmax -> q (bf16 into reset matrix)
//   MODE 1: storage gate   -> softmax weights ski (bf16)
//   MODE 2: detect signal  -> tanh(x + b) -> qk (f32)
// Block = 128 threads = 4 waves sharing one 64-column panel: TDM issued once
// (wave 0), consumed by everyone via ds_load_b128.
// ---------------------------------------------------------------------------
template <int MODE>
__global__ __launch_bounds__(128) void k_gemm(const bf16_t* __restrict__ A,
                                              const bf16_t* __restrict__ Wt,
                                              const float* __restrict__ bias,
                                              const float* __restrict__ state,
                                              bf16_t* __restrict__ reset_out,
                                              bf16_t* __restrict__ ski_out,
                                              float* __restrict__ qk_out,
                                              int nrows) {
  __shared__ __align__(16) unsigned char ldsB[2 * LDS_BUF];

  const int lane = threadIdx.x & 31;
  const int wave = threadIdx.x >> 5;
  const int r0   = blockIdx.y * 64 + wave * 16;
  const int n0   = blockIdx.x * 64;
  const int hi   = (lane < 16) ? 0 : 1;
  const int l15  = lane & 15;

  const v8f zero = {0.f, 0.f, 0.f, 0.f, 0.f, 0.f, 0.f, 0.f};
  v8f c0 = zero, c1 = zero, c2 = zero, c3 = zero;

  const bf16_t* wPanel = Wt + (size_t)n0 * KDIM;
  const int rowA = r0 + l15;
  const bf16_t* aRow = A + (size_t)rowA * KDIM + (hi ? 16 : 0);

  // Prologue: TDM stage k-step 0 into buffer 0; register-preload A frag 0.
  if (wave == 0) tdm_load_panel(wPanel, 0u, nrows);
  v8bf alo = *(const v8bf*)(aRow);
  v8bf ahi = *(const v8bf*)(aRow + 8);
  if (wave == 0) __builtin_amdgcn_s_wait_tensorcnt(0);
  __syncthreads();

  for (int ks = 0; ks < KSTEPS; ++ks) {
    const int cur = ks & 1;
    // Kick the TDM for the next k-step into the other buffer.
    if (wave == 0 && (ks + 1) < KSTEPS)
      tdm_load_panel(wPanel + (ks + 1) * 32, (unsigned)((cur ^ 1) * LDS_BUF),
                     nrows);
    // Register-pipeline the next A fragment (clamped on the last step).
    const int kn = ((ks + 1) < KSTEPS ? (ks + 1) : ks) * 32;
    const v8bf nlo = *(const v8bf*)(aRow + kn);
    const v8bf nhi = *(const v8bf*)(aRow + kn + 8);

    // Issue all eight B-fragment ds loads up front into distinct registers so
    // the scheduler can retire them with partial dscnt waits.
    const unsigned char* bbase = ldsB + cur * LDS_BUF + hi * 32;
    const v8bf* lp0 = (const v8bf*)(bbase + (0 * 16 + l15) * LDS_ROW);
    const v8bf* lp1 = (const v8bf*)(bbase + (1 * 16 + l15) * LDS_ROW);
    const v8bf* lp2 = (const v8bf*)(bbase + (2 * 16 + l15) * LDS_ROW);
    const v8bf* lp3 = (const v8bf*)(bbase + (3 * 16 + l15) * LDS_ROW);
    const v8bf b0l = lp0[0], b0h = lp0[1];
    const v8bf b1l = lp1[0], b1h = lp1[1];
    const v8bf b2l = lp2[0], b2h = lp2[1];
    const v8bf b3l = lp3[0], b3h = lp3[1];

    const v16bf af = pack16(alo, ahi);
    c0 = __builtin_amdgcn_wmma_f32_16x16x32_bf16(false, af, false,
                                                 pack16(b0l, b0h), (short)0,
                                                 c0, false, false);
    c1 = __builtin_amdgcn_wmma_f32_16x16x32_bf16(false, af, false,
                                                 pack16(b1l, b1h), (short)0,
                                                 c1, false, false);
    c2 = __builtin_amdgcn_wmma_f32_16x16x32_bf16(false, af, false,
                                                 pack16(b2l, b2h), (short)0,
                                                 c2, false, false);
    c3 = __builtin_amdgcn_wmma_f32_16x16x32_bf16(false, af, false,
                                                 pack16(b3l, b3h), (short)0,
                                                 c3, false, false);
    alo = nlo;
    ahi = nhi;
    if (wave == 0) __builtin_amdgcn_s_wait_tensorcnt(0);
    __syncthreads();
  }

  // Epilogue. C layout: lane = column (l15), VGPR i = row i (+8 upper half).
  v8f acc[4] = {c0, c1, c2, c3};
  const int rbase = r0 + (hi ? 8 : 0);
  #pragma unroll
  for (int nt = 0; nt < 4; ++nt) {
    const int col = n0 + nt * 16 + l15;
    const float bv = bias[col];
    const float tau = exp2f((float)(col & 7) * 1.66096404744f);  // 10^(m/2)
    #pragma unroll
    for (int i = 0; i < 8; ++i) {
      const int row = rbase + i;
      const float x = acc[nt][i] + bv;
      if (MODE == 2) {
        qk_out[(size_t)row * UNITS + col] = tanhf(x);
      } else {
        const float d = x - tau;
        float s = -d * d;
        float smax = s;
        smax = fmaxf(smax, __shfl_xor(smax, 1, 32));
        smax = fmaxf(smax, __shfl_xor(smax, 2, 32));
        smax = fmaxf(smax, __shfl_xor(smax, 4, 32));
        const float e = __expf(s - smax);
        float se = e;
        if (MODE == 0) {
          const float h = state[(size_t)row * NGATE + col];
          float seh = e * h;
          se += __shfl_xor(se, 1, 32);  seh += __shfl_xor(seh, 1, 32);
          se += __shfl_xor(se, 2, 32);  seh += __shfl_xor(seh, 2, 32);
          se += __shfl_xor(se, 4, 32);  seh += __shfl_xor(seh, 4, 32);
          if ((lane & 7) == 0)
            reset_out[(size_t)row * KDIM + D_IN + (col >> 3)] = (bf16_t)(seh / se);
        } else {
          se += __shfl_xor(se, 1, 32);
          se += __shfl_xor(se, 2, 32);
          se += __shfl_xor(se, 4, 32);
          ski_out[(size_t)row * NGATE + col] = (bf16_t)(e / se);
        }
      }
    }
  }
}

// ---------------------------------------------------------------------------
// h_hat_next = ((1-ski)*h_hat + ski*qk) * decay;  h_next = sum_m h_hat_next
// ---------------------------------------------------------------------------
__global__ void k_final(const float* __restrict__ state,
                        const bf16_t* __restrict__ ski,
                        const float* __restrict__ qk,
                        float* __restrict__ out_h,
                        float* __restrict__ out_hhat) {
  constexpr float DECAY[8] = {0.36787944f, 0.72889340f, 0.90483742f, 0.96887700f,
                              0.99004983f, 0.99684281f, 0.99900050f, 0.99968383f};
  const int idx = blockIdx.x * blockDim.x + threadIdx.x;  // [0, B*UNITS)
  const int b = idx >> 9;
  const int u = idx & 511;
  const size_t hoff = (size_t)b * NGATE + u * MBUCK;
  const float4* hp = (const float4*)(state + hoff);
  const float4 h0 = hp[0];
  const float4 h1 = hp[1];
  const v8bf sv = *(const v8bf*)(ski + hoff);
  const float q = qk[(size_t)b * UNITS + u];
  const float hh[8] = {h0.x, h0.y, h0.z, h0.w, h1.x, h1.y, h1.z, h1.w};
  float hn[8];
  float sum = 0.f;
  #pragma unroll
  for (int m = 0; m < 8; ++m) {
    const float s = (float)sv[m];
    const float v = ((1.0f - s) * hh[m] + s * q) * DECAY[m];
    hn[m] = v;
    sum += v;
  }
  out_h[(size_t)b * UNITS + u] = sum;
  float4* op = (float4*)(out_hhat + hoff);
  op[0] = make_float4(hn[0], hn[1], hn[2], hn[3]);
  op[1] = make_float4(hn[4], hn[5], hn[6], hn[7]);
}

// ---------------------------------------------------------------------------
extern "C" void kernel_launch(void* const* d_in, const int* in_sizes, int n_in,
                              void* d_out, int out_size, void* d_ws, size_t ws_size,
                              hipStream_t stream) {
  const float* inputs = (const float*)d_in[0];
  const float* state  = (const float*)d_in[1];
  const float* W_r    = (const float*)d_in[2];
  const float* b_r    = (const float*)d_in[3];
  const float* W_s    = (const float*)d_in[4];
  const float* b_s    = (const float*)d_in[5];
  const float* W_d    = (const float*)d_in[6];
  const float* b_d    = (const float*)d_in[7];

  float* out_h    = (float*)d_out;                      // B x 512
  float* out_hhat = out_h + (size_t)B_ROWS * UNITS;     // B x 4096

  char* ws = (char*)d_ws;
  size_t off = 0;
  auto carve = [&](size_t bytes) -> void* {
    void* p = (void*)(ws + off);
    off += (bytes + 255) & ~(size_t)255;
    return p;
  };
  bf16_t* Wr_t  = (bf16_t*)carve((size_t)NGATE * KDIM * 2);
  bf16_t* Ws_t  = (bf16_t*)carve((size_t)NGATE * KDIM * 2);
  bf16_t* Wd_t  = (bf16_t*)carve((size_t)UNITS * KDIM * 2);
  bf16_t* fused = (bf16_t*)carve((size_t)B_ROWS * KDIM * 2);
  bf16_t* reset = (bf16_t*)carve((size_t)B_ROWS * KDIM * 2);
  bf16_t* skib  = (bf16_t*)carve((size_t)B_ROWS * NGATE * 2);
  float*  qk    = (float*)carve((size_t)B_ROWS * UNITS * 4);

  const dim3 tb(32, 8);
  k_transpose_cvt<<<dim3(NGATE / 32, KDIM / 32), tb, 0, stream>>>(W_r, Wr_t, KDIM, NGATE);
  k_transpose_cvt<<<dim3(NGATE / 32, KDIM / 32), tb, 0, stream>>>(W_s, Ws_t, KDIM, NGATE);
  k_transpose_cvt<<<dim3(UNITS / 32, KDIM / 32), tb, 0, stream>>>(W_d, Wd_t, KDIM, UNITS);

  k_fuse<<<B_ROWS, 256, 0, stream>>>(inputs, state, fused, reset);

  // retrieval gate + q  (writes q columns of reset)
  k_gemm<0><<<dim3(NGATE / 64, B_ROWS / 64), 128, 0, stream>>>(
      fused, Wr_t, b_r, state, reset, nullptr, nullptr, NGATE);
  // storage gate -> ski
  k_gemm<1><<<dim3(NGATE / 64, B_ROWS / 64), 128, 0, stream>>>(
      fused, Ws_t, b_s, nullptr, nullptr, skib, nullptr, NGATE);
  // detect signal -> qk = tanh(reset @ W_d + b_d)
  k_gemm<2><<<dim3(UNITS / 64, B_ROWS / 64), 128, 0, stream>>>(
      reset, Wd_t, b_d, nullptr, nullptr, nullptr, qk, UNITS);

  k_final<<<(B_ROWS * UNITS) / 256, 256, 0, stream>>>(state, skib, qk, out_h, out_hhat);
}
